// DecoderOnlyAFT_13176959664295
// MI455X (gfx1250) — matrix operand order repaired and checked by reading, not compiled, not run on hardware
//
#include <hip/hip_runtime.h>
#include <cmath>

// ---- problem dims (fixed by reference) ----
#define LL    6
#define DD    512
#define HH    2048
#define TT    1024
#define BB    4
#define MM    (BB*TT)          // 4096 rows
#define WIN   64
#define LNEPS 1e-5f

typedef __attribute__((ext_vector_type(16))) _Float16 v16h;
typedef __attribute__((ext_vector_type(8)))  _Float16 h8;
typedef __attribute__((ext_vector_type(8)))  float    v8f;

#define WMMA16(a,b,c) __builtin_amdgcn_wmma_f32_16x16x32_f16(false, a, false, b, (short)0, c, false, false)

// ============================================================================
// WMMA GEMM: C[M,N] = A[M,K] @ W[K,N] (+bias, +relu).
// A: f16 row-major [M,K]; Wt: f16 TRANSPOSED [N,K]. f32 accum.
// Block: 256 threads (8 waves), 128x128 tile, double-buffered LDS.
// Waves: 4(M) x 2(N); each wave: 32x64 sub-tile = 2x4 WMMA tiles
//   -> 8 v_wmma per K-step per wave from 12 ds_load_b128.
// ============================================================================
template <int EPI>   // 0=none, 1=bias, 2=bias+relu
__global__ __launch_bounds__(256)
void gemm_wmma(const _Float16* __restrict__ A, const _Float16* __restrict__ Wt,
               const float* __restrict__ bias, float* __restrict__ C,
               int M, int N, int K)
{
    __shared__ alignas(16) _Float16 ldsA [2][128][40];  // [buf][m][k], 80B stride
    __shared__ alignas(16) _Float16 ldsBt[2][128][40];  // [buf][n][k]

    const int mBase = blockIdx.y * 128;
    const int nBase = blockIdx.x * 128;
    const int tid   = threadIdx.x;
    const int wave  = tid >> 5;
    const int lane  = tid & 31;
    const int half  = lane >> 4;
    const int lrow  = lane & 15;
    const int wm    = wave & 3;        // M sub-tile pair -> rows wm*32 .. +31
    const int wn    = wave >> 2;       // N half -> cols wn*64 .. +63
    const int m0    = wm * 32;
    const int n0    = wn * 64;

    // staging coords (fixed per thread): 2 chunks of 8 halves for A and B each
    const int sr = tid >> 2;           // 0..63
    const int sc = (tid & 3) * 8;      // 0,8,16,24

    const _Float16* gA0 = A  + (size_t)(mBase + sr)      * K + sc;
    const _Float16* gA1 = A  + (size_t)(mBase + sr + 64) * K + sc;
    const _Float16* gB0 = Wt + (size_t)(nBase + sr)      * K + sc;
    const _Float16* gB1 = Wt + (size_t)(nBase + sr + 64) * K + sc;

    v8f c00 = {0.f,0.f,0.f,0.f,0.f,0.f,0.f,0.f};
    v8f c01 = c00, c02 = c00, c03 = c00;
    v8f c10 = c00, c11 = c00, c12 = c00, c13 = c00;

    const int nsteps = K >> 5;

    // prologue stage into buffer 0
    *(h8*)&ldsA [0][sr     ][sc] = *(const h8*)gA0;  gA0 += 32;
    *(h8*)&ldsA [0][sr + 64][sc] = *(const h8*)gA1;  gA1 += 32;
    *(h8*)&ldsBt[0][sr     ][sc] = *(const h8*)gB0;  gB0 += 32;
    *(h8*)&ldsBt[0][sr + 64][sc] = *(const h8*)gB1;  gB1 += 32;
    __syncthreads();

    for (int s = 0; s < nsteps; ++s) {
        const int cur = s & 1;
        if (s + 1 < nsteps) {
            const int nxt = cur ^ 1;
            *(h8*)&ldsA [nxt][sr     ][sc] = *(const h8*)gA0;  gA0 += 32;
            *(h8*)&ldsA [nxt][sr + 64][sc] = *(const h8*)gA1;  gA1 += 32;
            *(h8*)&ldsBt[nxt][sr     ][sc] = *(const h8*)gB0;  gB0 += 32;
            *(h8*)&ldsBt[nxt][sr + 64][sc] = *(const h8*)gB1;  gB1 += 32;
        }

        // A fragments (two m-tiles): two contiguous 8-half runs per lane each
        const h8 a0lo = *(const h8*)&ldsA[cur][m0 + lrow     ][half * 8];
        const h8 a0hi = *(const h8*)&ldsA[cur][m0 + lrow     ][16 + half * 8];
        const h8 a1lo = *(const h8*)&ldsA[cur][m0 + 16 + lrow][half * 8];
        const h8 a1hi = *(const h8*)&ldsA[cur][m0 + 16 + lrow][16 + half * 8];
        const v16h a0 = __builtin_shufflevector(a0lo, a0hi, 0,1,2,3,4,5,6,7,8,9,10,11,12,13,14,15);
        const v16h a1 = __builtin_shufflevector(a1lo, a1hi, 0,1,2,3,4,5,6,7,8,9,10,11,12,13,14,15);

        #pragma unroll
        for (int nt = 0; nt < 4; ++nt) {
            const int n = n0 + nt * 16 + lrow;
            const h8 blo = *(const h8*)&ldsBt[cur][n][half * 8];
            const h8 bhi = *(const h8*)&ldsBt[cur][n][16 + half * 8];
            const v16h b = __builtin_shufflevector(blo, bhi, 0,1,2,3,4,5,6,7,8,9,10,11,12,13,14,15);
            if      (nt == 0) { c00 = WMMA16(a0, b, c00); c10 = WMMA16(a1, b, c10); }
            else if (nt == 1) { c01 = WMMA16(a0, b, c01); c11 = WMMA16(a1, b, c11); }
            else if (nt == 2) { c02 = WMMA16(a0, b, c02); c12 = WMMA16(a1, b, c12); }
            else              { c03 = WMMA16(a0, b, c03); c13 = WMMA16(a1, b, c13); }
        }
        __syncthreads();
    }

    // epilogue. C/D layout: vgpr r -> row = r + 8*half, col = lane&15
    #pragma unroll
    for (int nt = 0; nt < 4; ++nt) {
        const int col = nBase + n0 + nt * 16 + lrow;
        float bv = 0.f;
        if (EPI >= 1) bv = bias[col];
        const v8f accA = (nt == 0) ? c00 : (nt == 1) ? c01 : (nt == 2) ? c02 : c03;
        const v8f accB = (nt == 0) ? c10 : (nt == 1) ? c11 : (nt == 2) ? c12 : c13;
        #pragma unroll
        for (int r = 0; r < 8; ++r) {
            int row = mBase + m0 + half * 8 + r;
            float o0 = accA[r] + bv;
            float o1 = accB[r] + bv;
            if (EPI == 2) { o0 = fmaxf(o0, 0.f); o1 = fmaxf(o1, 0.f); }
            C[(size_t)row * N + col]        = o0;
            C[(size_t)(row + 16) * N + col] = o1;
        }
    }
}

// ============================================================================
// f32 [K,N] -> f16 transposed [N,K] (LDS-tiled)
// ============================================================================
__global__ __launch_bounds__(256)
void transpose_cvt_k(_Float16* __restrict__ Wt, const float* __restrict__ W,
                     int K, int N)
{
    __shared__ float tile[32][33];
    const int n0 = blockIdx.x * 32;
    const int k0 = blockIdx.y * 32;
    const int tx = threadIdx.x & 31;
    const int ty = threadIdx.x >> 5;       // 0..7
    #pragma unroll
    for (int j = 0; j < 4; ++j)
        tile[ty + j * 8][tx] = W[(size_t)(k0 + ty + j * 8) * N + n0 + tx];
    __syncthreads();
    #pragma unroll
    for (int j = 0; j < 4; ++j)
        Wt[(size_t)(n0 + ty + j * 8) * K + k0 + tx] = (_Float16)tile[tx][ty + j * 8];
}

// f32 -> f16 elementwise, 8 per thread (b128 in, b128 out)
__global__ __launch_bounds__(256)
void cvt_f16_k(_Float16* __restrict__ o, const float* __restrict__ in)
{
    size_t i = ((size_t)blockIdx.x * 256 + threadIdx.x) * 8;
    const float4* src = (const float4*)&in[i];
    float4 p0 = src[0], p1 = src[1];
    h8 hv = { (_Float16)p0.x, (_Float16)p0.y, (_Float16)p0.z, (_Float16)p0.w,
              (_Float16)p1.x, (_Float16)p1.y, (_Float16)p1.z, (_Float16)p1.w };
    *(h8*)&o[i] = hv;
}

// ============================================================================
// Elementwise / reduction kernels
// ============================================================================
__global__ void pos_scale_k(float* __restrict__ ps, const float* __restrict__ pos, float scale)
{
    int i = blockIdx.x * 256 + threadIdx.x;
    ps[i] = pos[i] * scale;
}

__global__ void embed_k(float* __restrict__ h, const int* __restrict__ x,
                        const float* __restrict__ tok, const float* __restrict__ ps, float scale)
{
    int i  = blockIdx.x * 256 + threadIdx.x;
    int d  = i & (DD - 1);
    int bt = i >> 9;
    int t  = bt & (TT - 1);
    int tk = x[bt];
    h[i] = tok[(size_t)tk * DD + d] * scale + ps[t * DD + d];
}

__global__ void add_pos_k(float* __restrict__ emb, const float* __restrict__ h,
                          const float* __restrict__ ps)
{
    int i = blockIdx.x * 256 + threadIdx.x;
    emb[i] = h[i] + ps[i & (TT * DD - 1)];
}

__global__ void kmax_k(float* __restrict__ kmax, const float* __restrict__ k)
{
    int i = blockIdx.x * 256 + threadIdx.x;          // B*D threads
    int b = i >> 9, d = i & (DD - 1);
    float m = -1e30f;
    for (int t = 0; t < TT; ++t)
        m = fmaxf(m, k[((size_t)b * TT + t) * DD + d]);
    kmax[i] = m;
}

__global__ void ekv_k(float* __restrict__ k, float* __restrict__ v,
                      const float* __restrict__ kmax)
{
    int i = blockIdx.x * 256 + threadIdx.x;
    int d = i & (DD - 1);
    int b = i / (TT * DD);
    float e = __expf(k[i] - kmax[b * DD + d]);
    k[i] = e;                 // ek  (in place)
    v[i] = e * v[i];          // ek*v (in place)
}

// ew[t][j], j in [0,64): u = t-63+j; row-stable exp over the causal local window
__global__ void ewband_k(float* __restrict__ ew, const float* __restrict__ wb)
{
    __shared__ float s[64];
    int t = blockIdx.x, j = threadIdx.x;
    int u = t - (WIN - 1) + j;
    float wm = (u >= 0) ? wb[(size_t)t * TT + u] : -1e30f;
    s[j] = wm;
    __syncthreads();
    for (int st = 32; st > 0; st >>= 1) {
        if (j < st) s[j] = fmaxf(s[j], s[j + st]);
        __syncthreads();
    }
    float rmax = s[0];
    ew[t * WIN + j] = (u >= 0) ? __expf(wm - rmax) : 0.f;
}

// num/den[b,t,:] = sum_{u in window} ew[t,u] * {ekv,ek}[b,u,:]
__global__ __launch_bounds__(512)
void banded_k(float* __restrict__ num, float* __restrict__ den,
              const float* __restrict__ ew, const float* __restrict__ ek,
              const float* __restrict__ ekv)
{
    __shared__ float sew[WIN];
    int t = blockIdx.x, b = blockIdx.y, d = threadIdx.x;
    if (d < WIN) sew[d] = ew[t * WIN + d];
    __syncthreads();
    int u0 = t - (WIN - 1);
    int j0 = (u0 < 0) ? -u0 : 0;
    const size_t base = (size_t)b * TT * DD + d;
    float na = 0.f, da = 0.f;
    for (int j = j0; j < WIN; ++j) {
        float w = sew[j];
        size_t idx = base + (size_t)(u0 + j) * DD;
        na = fmaf(w, ekv[idx], na);
        da = fmaf(w, ek[idx], da);
    }
    size_t o = ((size_t)b * TT + t) * DD + d;
    num[o] = na;
    den[o] = da;
}

__global__ void y_k(float* __restrict__ q, const float* __restrict__ num,
                    const float* __restrict__ den)
{
    int i = blockIdx.x * 256 + threadIdx.x;
    float s = 1.f / (1.f + __expf(-q[i]));
    q[i] = s * (num[i] / (den[i] + 1e-9f));
}

// out = LayerNorm(a + res) over D=512; block per row
__global__ __launch_bounds__(256)
void addln_k(float* __restrict__ out, const float* __restrict__ a,
             const float* __restrict__ res, const float* __restrict__ g,
             const float* __restrict__ bb)
{
    __shared__ float red[256];
    int row = blockIdx.x, tid = threadIdx.x;
    size_t base = (size_t)row * DD;
    float x0 = a[base + tid]       + res[base + tid];
    float x1 = a[base + tid + 256] + res[base + tid + 256];
    red[tid] = x0 + x1;
    __syncthreads();
    for (int st = 128; st > 0; st >>= 1) {
        if (tid < st) red[tid] += red[tid + st];
        __syncthreads();
    }
    float mean = red[0] * (1.f / DD);
    __syncthreads();
    float d0 = x0 - mean, d1 = x1 - mean;
    red[tid] = d0 * d0 + d1 * d1;
    __syncthreads();
    for (int st = 128; st > 0; st >>= 1) {
        if (tid < st) red[tid] += red[tid + st];
        __syncthreads();
    }
    float rstd = rsqrtf(red[0] * (1.f / DD) + LNEPS);
    out[base + tid]       = d0 * rstd * g[tid]       + bb[tid];
    out[base + tid + 256] = d1 * rstd * g[tid + 256] + bb[tid + 256];
}

// ============================================================================
// Driver
// ============================================================================
extern "C" void kernel_launch(void* const* d_in, const int* in_sizes, int n_in,
                              void* d_out, int out_size, void* d_ws, size_t ws_size,
                              hipStream_t stream)
{
    const int*   x   = (const int*)  d_in[0];
    const float* tok = (const float*)d_in[1];
    const float* pos = (const float*)d_in[2];
    const float* Wq  = (const float*)d_in[3];
    const float* bq  = (const float*)d_in[4];
    const float* Wk  = (const float*)d_in[5];
    const float* bk  = (const float*)d_in[6];
    const float* Wv  = (const float*)d_in[7];
    const float* bv  = (const float*)d_in[8];
    const float* Wo  = (const float*)d_in[9];
    const float* bo  = (const float*)d_in[10];
    const float* wb  = (const float*)d_in[11];
    const float* lng = (const float*)d_in[12];
    const float* lnb = (const float*)d_in[13];
    const float* W1  = (const float*)d_in[14];
    const float* b1  = (const float*)d_in[15];
    const float* W2  = (const float*)d_in[16];
    const float* b2  = (const float*)d_in[17];

    float* h = (float*)d_out;                        // [B,T,D] fp32

    // workspace carve-out (byte-based; 256B alignment)
    char* wsb = (char*)d_ws;
    size_t off = 0;
    auto carveB = [&](size_t bytes) { void* p = wsb + off; off = (off + bytes + 255) & ~(size_t)255; return p; };
    float* ps    = (float*)carveB(sizeof(float) * TT * DD);
    float* emb   = (float*)carveB(sizeof(float) * MM * DD);
    float* q     = (float*)carveB(sizeof(float) * MM * DD);      // later: y (in place)
    float* kbuf  = (float*)carveB(sizeof(float) * MM * DD);      // k -> ek -> f
    float* vbuf  = (float*)carveB(sizeof(float) * MM * DD);      // v -> ek*v
    float* num   = (float*)carveB(sizeof(float) * MM * DD);      // later: a
    float* den   = (float*)carveB(sizeof(float) * MM * DD);      // later: attn_out
    float* f1    = (float*)carveB(sizeof(float) * MM * HH);
    float* kmax  = (float*)carveB(sizeof(float) * BB * DD);
    float* ew    = (float*)carveB(sizeof(float) * TT * WIN);
    _Float16* embh = (_Float16*)carveB(sizeof(_Float16) * MM * DD);
    _Float16* actH = (_Float16*)carveB(sizeof(_Float16) * MM * HH);
    _Float16* Wqt  = (_Float16*)carveB(sizeof(_Float16) * LL * DD * DD);
    _Float16* Wkt  = (_Float16*)carveB(sizeof(_Float16) * LL * DD * DD);
    _Float16* Wvt  = (_Float16*)carveB(sizeof(_Float16) * LL * DD * DD);
    _Float16* Wot  = (_Float16*)carveB(sizeof(_Float16) * LL * DD * DD);
    _Float16* W1t  = (_Float16*)carveB(sizeof(_Float16) * LL * DD * HH);  // [H][D] per layer
    _Float16* W2t  = (_Float16*)carveB(sizeof(_Float16) * LL * HH * DD);  // [D][H] per layer

    const float scale = 1.0f / sqrtf((float)DD);
    const int nBTD = MM * DD;

    pos_scale_k<<<(TT * DD) / 256, 256, 0, stream>>>(ps, pos, scale);
    embed_k<<<nBTD / 256, 256, 0, stream>>>(h, x, tok, ps, scale);

    // pre-transpose + convert all weights to f16 [N][K]
    dim3 tDD(DD / 32, DD / 32);          // K=D, N=D
    dim3 tDH(HH / 32, DD / 32);          // W1: K=D, N=H
    dim3 tHD(DD / 32, HH / 32);          // W2: K=H, N=D
    for (int l = 0; l < LL; ++l) {
        transpose_cvt_k<<<tDD, 256, 0, stream>>>(Wqt + (size_t)l * DD * DD, Wq + (size_t)l * DD * DD, DD, DD);
        transpose_cvt_k<<<tDD, 256, 0, stream>>>(Wkt + (size_t)l * DD * DD, Wk + (size_t)l * DD * DD, DD, DD);
        transpose_cvt_k<<<tDD, 256, 0, stream>>>(Wvt + (size_t)l * DD * DD, Wv + (size_t)l * DD * DD, DD, DD);
        transpose_cvt_k<<<tDD, 256, 0, stream>>>(Wot + (size_t)l * DD * DD, Wo + (size_t)l * DD * DD, DD, DD);
        transpose_cvt_k<<<tDH, 256, 0, stream>>>(W1t + (size_t)l * DD * HH, W1 + (size_t)l * DD * HH, DD, HH);
        transpose_cvt_k<<<tHD, 256, 0, stream>>>(W2t + (size_t)l * HH * DD, W2 + (size_t)l * HH * DD, HH, DD);
    }

    dim3 gD(DD / 128, MM / 128);   // 4 x 32 blocks
    dim3 gH(HH / 128, MM / 128);   // 16 x 32 blocks

    for (int l = 0; l < LL; ++l) {
        const _Float16* Wqt_l = Wqt + (size_t)l * DD * DD;
        const _Float16* Wkt_l = Wkt + (size_t)l * DD * DD;
        const _Float16* Wvt_l = Wvt + (size_t)l * DD * DD;
        const _Float16* Wot_l = Wot + (size_t)l * DD * DD;
        const _Float16* W1t_l = W1t + (size_t)l * DD * HH;
        const _Float16* W2t_l = W2t + (size_t)l * HH * DD;
        const float* bq_l = bq + (size_t)l * DD;
        const float* bk_l = bk + (size_t)l * DD;
        const float* bv_l = bv + (size_t)l * DD;
        const float* bo_l = bo + (size_t)l * DD;
        const float* b1_l = b1 + (size_t)l * HH;
        const float* b2_l = b2 + (size_t)l * DD;
        const float* g_l  = lng + (size_t)l * DD;
        const float* be_l = lnb + (size_t)l * DD;
        const float* wb_l = wb + (size_t)l * TT * TT;

        add_pos_k<<<nBTD / 256, 256, 0, stream>>>(emb, h, ps);
        cvt_f16_k<<<nBTD / 2048, 256, 0, stream>>>(embh, emb);

        gemm_wmma<1><<<gD, 256, 0, stream>>>(embh, Wqt_l, bq_l, q,    MM, DD, DD);
        gemm_wmma<1><<<gD, 256, 0, stream>>>(embh, Wkt_l, bk_l, kbuf, MM, DD, DD);
        gemm_wmma<1><<<gD, 256, 0, stream>>>(embh, Wvt_l, bv_l, vbuf, MM, DD, DD);

        kmax_k<<<(BB * DD) / 256, 256, 0, stream>>>(kmax, kbuf);
        ekv_k<<<nBTD / 256, 256, 0, stream>>>(kbuf, vbuf, kmax);
        ewband_k<<<TT, WIN, 0, stream>>>(ew, wb_l);
        banded_k<<<dim3(TT, BB), 512, 0, stream>>>(num, den, ew, kbuf, vbuf);
        y_k<<<nBTD / 256, 256, 0, stream>>>(q, num, den);

        cvt_f16_k<<<nBTD / 2048, 256, 0, stream>>>(actH, q);
        gemm_wmma<1><<<gD, 256, 0, stream>>>(actH, Wot_l, bo_l, num, MM, DD, DD);  // a -> num
        addln_k<<<MM, 256, 0, stream>>>(den, num, emb, g_l, be_l);                 // attn_out -> den

        cvt_f16_k<<<nBTD / 2048, 256, 0, stream>>>(actH, den);
        gemm_wmma<2><<<gH, 256, 0, stream>>>(actH, W1t_l, b1_l, f1, MM, HH, DD);   // relu
        cvt_f16_k<<<(MM * HH) / 2048, 256, 0, stream>>>(actH, f1);
        gemm_wmma<1><<<gD, 256, 0, stream>>>(actH, W2t_l, b2_l, kbuf, MM, DD, HH); // f -> kbuf
        addln_k<<<MM, 256, 0, stream>>>(h, kbuf, den, g_l, be_l);
    }
    (void)in_sizes; (void)n_in; (void)out_size; (void)ws_size;
}
